// TernaryMVAdapter_65841848648288
// MI455X (gfx1250) — 8-bit matrix operand order repaired and checked by reading, compiled, not run on hardware
//
#include <hip/hip_runtime.h>
#include <cstdint>
#include <cstddef>

#define CDIV(a, b) (((a) + (b) - 1) / (b))

typedef int v8i __attribute__((ext_vector_type(8)));

__device__ __forceinline__ float siluf(float x) { return x / (1.f + expf(-x)); }
__device__ __forceinline__ float sigmf(float x) { return 1.f / (1.f + expf(-x)); }

__device__ __forceinline__ void async_ld_b128(void* lds, const void* gptr) {
  asm volatile("global_load_async_to_lds_b128 %0, %1, off"
               :: "v"((unsigned)(size_t)lds),
                  "v"((unsigned long long)(size_t)gptr)
               : "memory");
}
__device__ __forceinline__ void wait_async_le3() {
  asm volatile("s_wait_asynccnt 0x3" ::: "memory");
}
__device__ __forceinline__ void wait_async_0() {
  asm volatile("s_wait_asynccnt 0x0" ::: "memory");
}

__global__ __launch_bounds__(1024) void k_absmean(const float* __restrict__ w,
                                                  long n, float* __restrict__ wdq) {
  __shared__ float red[1024];
  float s = 0.f;
  for (long i = threadIdx.x; i < n; i += 1024) s += fabsf(w[i]);
  red[threadIdx.x] = s;
  __syncthreads();
  for (int st = 512; st > 0; st >>= 1) {
    if ((int)threadIdx.x < st) red[threadIdx.x] += red[threadIdx.x + st];
    __syncthreads();
  }
  if (threadIdx.x == 0) wdq[0] = fmaxf(red[0] / (float)n, 1e-5f);
}

__global__ void k_ternarize(const float* __restrict__ w, const float* __restrict__ wdq,
                            int8_t* __restrict__ q, long n) {
  long i = (long)blockIdx.x * 256 + threadIdx.x;
  if (i >= n) return;
  float s = 1.f / wdq[0];
  float v = rintf(w[i] * s);
  v = fminf(1.f, fmaxf(-1.f, v));
  q[i] = (int8_t)v;
}

__global__ __launch_bounds__(256) void k_rmsnorm_quant(const float* __restrict__ X,
                                                       const float* __restrict__ g,
                                                       int8_t* __restrict__ Q,
                                                       float* __restrict__ adq,
                                                       int M, int K) {
  int row = blockIdx.x;
  if (row >= M) return;
  const float* x = X + (size_t)row * K;
  __shared__ float red[256];
  int tid = threadIdx.x;
  float ss = 0.f;
  for (int k = tid; k < K; k += 256) { float v = x[k]; ss += v * v; }
  red[tid] = ss; __syncthreads();
  for (int st = 128; st > 0; st >>= 1) { if (tid < st) red[tid] += red[tid + st]; __syncthreads(); }
  float rms = rsqrtf(red[0] / (float)K + 1e-6f);
  __syncthreads();
  float am = 0.f;
  for (int k = tid; k < K; k += 256) am = fmaxf(am, fabsf(x[k] * rms * g[k]));
  red[tid] = am; __syncthreads();
  for (int st = 128; st > 0; st >>= 1) { if (tid < st) red[tid] = fmaxf(red[tid], red[tid + st]); __syncthreads(); }
  float amax = fmaxf(red[0], 1e-5f);
  float qs = 127.f / amax;
  for (int k = tid; k < K; k += 256) {
    float y = x[k] * rms * g[k];
    int qi = (int)rintf(y * qs);
    qi = min(127, max(-128, qi));
    Q[(size_t)row * K + k] = (int8_t)qi;
  }
  if (tid == 0) adq[row] = amax / 127.f;
}

__global__ __launch_bounds__(256) void k_gemm_iu8_fast(const int8_t* __restrict__ A,
                                                       const float* __restrict__ adq,
                                                       const int8_t* __restrict__ B,
                                                       const float* __restrict__ wdq,
                                                       const float* __restrict__ bias,
                                                       float* __restrict__ C,
                                                       int M, int N, int K) {
  constexpr int LDA = 80, LDB = 80;
  __shared__ __align__(16) int8_t lA[2][64 * LDA];
  __shared__ __align__(16) int8_t lB[2][128 * LDB];
  const int tid = threadIdx.x;
  const int bm = blockIdx.y * 64;
  const int bn = blockIdx.x * 128;
  const int wid = tid >> 5, lane = tid & 31;
  const int wm = (wid >> 1) * 16;
  const int wn = (wid & 1) * 64;
  const int lrow = lane & 15, hi = lane >> 4;
  const int ar = tid >> 2, ac = (tid & 3) * 16;
  const int br = tid >> 1, bc = (tid & 1) * 32;

  const int8_t* Ap = A + (size_t)(bm + ar) * K + ac;
  const int8_t* Bp = B + (size_t)(bn + br) * K + bc;

  v8i zero = {0, 0, 0, 0, 0, 0, 0, 0};
  v8i acc[4];
#pragma unroll
  for (int s = 0; s < 4; ++s) acc[s] = zero;

  async_ld_b128(lA[0] + ar * LDA + ac, Ap);
  async_ld_b128(lB[0] + br * LDB + bc, Bp);
  async_ld_b128(lB[0] + br * LDB + bc + 16, Bp + 16);

  int buf = 0;
  for (int k0 = 0; k0 < K; k0 += 64) {
    if (k0 + 64 < K) {
      const int8_t* An = Ap + k0 + 64;
      const int8_t* Bn = Bp + k0 + 64;
      async_ld_b128(lA[buf ^ 1] + ar * LDA + ac, An);
      async_ld_b128(lB[buf ^ 1] + br * LDB + bc, Bn);
      async_ld_b128(lB[buf ^ 1] + br * LDB + bc + 16, Bn + 16);
      wait_async_le3();
    } else {
      wait_async_0();
    }
    __syncthreads();

    const int8_t* cA = lA[buf];
    const int8_t* cB = lB[buf];
    v8i af;
#pragma unroll
    for (int p = 0; p < 2; ++p) {
      int4 d = *(const int4*)(cA + (wm + lrow) * LDA + p * 32 + hi * 16);
      af[4 * p] = d.x;
      af[4 * p + 1] = d.y;
      af[4 * p + 2] = d.z;
      af[4 * p + 3] = d.w;
    }
#pragma unroll
    for (int s = 0; s < 4; ++s) {
      const int8_t* r = cB + (wn + s * 16 + lrow) * LDB + hi * 16;
      int4 lo = *(const int4*)(r);
      int4 hh = *(const int4*)(r + 32);
      v8i bf;
      bf[0] = lo.x; bf[1] = lo.y; bf[2] = lo.z; bf[3] = lo.w;
      bf[4] = hh.x; bf[5] = hh.y; bf[6] = hh.z; bf[7] = hh.w;
      acc[s] = __builtin_amdgcn_wmma_i32_16x16x64_iu8(true, af, true, bf, acc[s], false, false);
    }
    __syncthreads();
    buf ^= 1;
  }

  const float wsv = wdq[0];
#pragma unroll
  for (int s = 0; s < 4; ++s) {
    int n0 = bn + wn + s * 16 + lrow;
    float bv = bias ? bias[n0] : 0.f;
#pragma unroll
    for (int i = 0; i < 8; ++i) {
      int m = bm + wm + i + hi * 8;
      C[(size_t)m * N + n0] = (float)acc[s][i] * (adq[m] * wsv) + bv;
    }
  }
}

__global__ __launch_bounds__(256) void k_gemm_iu8(const int8_t* __restrict__ A,
                                                  const float* __restrict__ adq,
                                                  const int8_t* __restrict__ B,
                                                  const float* __restrict__ wdq,
                                                  const float* __restrict__ bias,
                                                  float* __restrict__ C,
                                                  int M, int N, int K) {
  constexpr int LDA = 80, LDB = 80;
  __shared__ __align__(16) int8_t lA[64 * LDA];
  __shared__ __align__(16) int8_t lB[64 * LDB];
  const int tid = threadIdx.x;
  const int bm = blockIdx.y * 64;
  const int bn = blockIdx.x * 64;
  const int wid = tid >> 5, lane = tid & 31;
  const int wm = (wid >> 1) * 16;
  const int wn = (wid & 1) * 32;
  const int lrow = lane & 15, hi = lane >> 4;
  const int ldr = tid >> 2, ldc = (tid & 3) * 16;

  v8i acc0 = {0, 0, 0, 0, 0, 0, 0, 0};
  v8i acc1 = {0, 0, 0, 0, 0, 0, 0, 0};

  for (int k0 = 0; k0 < K; k0 += 64) {
    int4 va = make_int4(0, 0, 0, 0), vb = make_int4(0, 0, 0, 0);
    const int gm = bm + ldr, gn = bn + ldr, gk = k0 + ldc;
    if (gm < M && gk < K) va = *(const int4*)(A + (size_t)gm * K + gk);
    if (gn < N && gk < K) vb = *(const int4*)(B + (size_t)gn * K + gk);
    *(int4*)(lA + ldr * LDA + ldc) = va;
    *(int4*)(lB + ldr * LDB + ldc) = vb;
    if (k0 + 64 < K) {
      if (gm < M) __builtin_prefetch((const void*)(A + (size_t)gm * K + gk + 64), 0, 3);
      if (gn < N) __builtin_prefetch((const void*)(B + (size_t)gn * K + gk + 64), 0, 3);
    }
    __syncthreads();

    v8i af;
#pragma unroll
    for (int p = 0; p < 2; ++p) {
      int4 d = *(const int4*)(lA + (wm + lrow) * LDA + p * 32 + hi * 16);
      af[4 * p] = d.x;
      af[4 * p + 1] = d.y;
      af[4 * p + 2] = d.z;
      af[4 * p + 3] = d.w;
    }
    v8i bf0, bf1;
    {
      const int8_t* r0 = lB + (wn + lrow) * LDB + hi * 16;
      int4 lo = *(const int4*)(r0);
      int4 hh = *(const int4*)(r0 + 32);
      bf0[0] = lo.x; bf0[1] = lo.y; bf0[2] = lo.z; bf0[3] = lo.w;
      bf0[4] = hh.x; bf0[5] = hh.y; bf0[6] = hh.z; bf0[7] = hh.w;
      const int8_t* r1 = lB + (wn + 16 + lrow) * LDB + hi * 16;
      lo = *(const int4*)(r1);
      hh = *(const int4*)(r1 + 32);
      bf1[0] = lo.x; bf1[1] = lo.y; bf1[2] = lo.z; bf1[3] = lo.w;
      bf1[4] = hh.x; bf1[5] = hh.y; bf1[6] = hh.z; bf1[7] = hh.w;
    }
    acc0 = __builtin_amdgcn_wmma_i32_16x16x64_iu8(true, af, true, bf0, acc0, false, false);
    acc1 = __builtin_amdgcn_wmma_i32_16x16x64_iu8(true, af, true, bf1, acc1, false, false);
    __syncthreads();
  }

  const float wsv = wdq[0];
#pragma unroll
  for (int i = 0; i < 8; ++i) {
    int m = bm + wm + i + hi * 8;
    if (m >= M) continue;
    float as = adq[m] * wsv;
    int n0 = bn + wn + lrow;
    if (n0 < N) C[(size_t)m * N + n0] = (float)acc0[i] * as + (bias ? bias[n0] : 0.f);
    int n1 = n0 + 16;
    if (n1 < N) C[(size_t)m * N + n1] = (float)acc1[i] * as + (bias ? bias[n1] : 0.f);
  }
}

__global__ void k_gemm_f32(const float* __restrict__ X, const float* __restrict__ W,
                           const float* __restrict__ bias, float* __restrict__ Y,
                           int M, int N, int K) {
  long idx = (long)blockIdx.x * 256 + threadIdx.x;
  if (idx >= (long)M * N) return;
  int m = (int)(idx / N), n = (int)(idx % N);
  const float* xr = X + (size_t)m * K;
  const float* wr = W + (size_t)n * K;
  float s = bias ? bias[n] : 0.f;
  for (int k = 0; k < K; ++k) s += xr[k] * wr[k];
  Y[idx] = s;
}

__global__ __launch_bounds__(256) void k_lnmod(const float* __restrict__ X,
                                               float* __restrict__ Y, int M, int D, int L,
                                               const float* __restrict__ mod, int mstride,
                                               int sh_off, int sc_off,
                                               const float* __restrict__ w,
                                               const float* __restrict__ b, float eps) {
  int row = blockIdx.x;
  if (row >= M) return;
  const float* x = X + (size_t)row * D;
  __shared__ float red[256];
  int tid = threadIdx.x;
  float s = 0.f;
  for (int d = tid; d < D; d += 256) s += x[d];
  red[tid] = s; __syncthreads();
  for (int st = 128; st > 0; st >>= 1) { if (tid < st) red[tid] += red[tid + st]; __syncthreads(); }
  float mean = red[0] / (float)D;
  __syncthreads();
  float v = 0.f;
  for (int d = tid; d < D; d += 256) { float t = x[d] - mean; v += t * t; }
  red[tid] = v; __syncthreads();
  for (int st = 128; st > 0; st >>= 1) { if (tid < st) red[tid] += red[tid + st]; __syncthreads(); }
  float inv = rsqrtf(red[0] / (float)D + eps);
  int bn = row / L;
  for (int d = tid; d < D; d += 256) {
    float y = (x[d] - mean) * inv;
    if (w) y = y * w[d] + b[d];
    if (mod) y = y * (1.f + mod[(size_t)bn * mstride + sc_off + d]) +
                 mod[(size_t)bn * mstride + sh_off + d];
    Y[(size_t)row * D + d] = y;
  }
}

__global__ void k_silu(const float* __restrict__ x, float* __restrict__ y, long n) {
  long i = (long)blockIdx.x * 256 + threadIdx.x;
  if (i < n) y[i] = siluf(x[i]);
}
__global__ void k_addinp(float* __restrict__ a, const float* __restrict__ b, long n) {
  long i = (long)blockIdx.x * 256 + threadIdx.x;
  if (i < n) a[i] += b[i];
}
__global__ void k_gate(float* __restrict__ iraw, float* __restrict__ fraw, long n) {
  long i = (long)blockIdx.x * 256 + threadIdx.x;
  if (i >= n) return;
  float f = sigmf(fraw[i]);
  float si = siluf(iraw[i]);
  fraw[i] = f;
  iraw[i] = si * (1.f - f);
}
__global__ void k_scan(const float* __restrict__ f, const float* __restrict__ in,
                       float* __restrict__ o, int Bn, int L, int D) {
  long idx = (long)blockIdx.x * 256 + threadIdx.x;
  if (idx >= (long)Bn * D) return;
  int b = (int)(idx / D), d = (int)(idx % D);
  size_t base = (size_t)b * L * D + d;
  float h = 0.f;
  for (int l = 0; l < L; ++l) {
    size_t p = base + (size_t)l * D;
    h = f[p] * h + in[p];
    o[p] = h;
  }
}
__global__ void k_headgate(const float* __restrict__ o, const float* __restrict__ g,
                           const float* __restrict__ gn, float* __restrict__ y,
                           int M, int NH, int HD) {
  long idx = (long)blockIdx.x * 256 + threadIdx.x;
  if (idx >= (long)M * NH) return;
  int m = (int)(idx / NH), h = (int)(idx % NH);
  const float* op = o + (size_t)m * NH * HD + (size_t)h * HD;
  const float* gp = g + (size_t)m * NH * HD + (size_t)h * HD;
  float ss = 0.f;
  for (int d = 0; d < HD; ++d) { float v = op[d]; ss += v * v; }
  float rms = rsqrtf(ss / (float)HD + 1e-6f);
  float* yp = y + (size_t)m * NH * HD + (size_t)h * HD;
  for (int d = 0; d < HD; ++d) yp[d] = op[d] * rms * gn[d] * siluf(gp[d]);
}
__global__ void k_resmod(float* __restrict__ X, const float* __restrict__ A,
                         const float* __restrict__ mod, int mstride, int off,
                         int M, int D, int L) {
  long idx = (long)blockIdx.x * 256 + threadIdx.x;
  if (idx >= (long)M * D) return;
  int row = (int)(idx / D), d = (int)(idx % D);
  int bn = row / L;
  X[idx] += mod[(size_t)bn * mstride + off + d] * A[idx];
}
__global__ void k_swiglu(const float* __restrict__ G, float* __restrict__ y,
                         int M, int inter) {
  long idx = (long)blockIdx.x * 256 + threadIdx.x;
  if (idx >= (long)M * inter) return;
  int m = (int)(idx / inter), j = (int)(idx % inter);
  float gv = G[(size_t)m * 2 * inter + j];
  float yv = G[(size_t)m * 2 * inter + inter + j];
  y[idx] = siluf(gv) * yv;
}
__global__ void k_tfreq(const float* __restrict__ t, float* __restrict__ out, int BN) {
  int idx = blockIdx.x * 256 + threadIdx.x;
  if (idx >= BN * 128) return;
  int bn = idx / 128, j = idx % 128;
  float fr = expf(-logf(10000.f) * (float)j / 128.f);
  float ang = t[bn] * fr;
  out[(size_t)bn * 256 + j] = cosf(ang);
  out[(size_t)bn * 256 + 128 + j] = sinf(ang);
}
__global__ void k_cond(const float* __restrict__ enc, const float* __restrict__ temb,
                       float* __restrict__ c, int BN, int NV, int S, int D) {
  long idx = (long)blockIdx.x * 256 + threadIdx.x;
  if (idx >= (long)BN * D) return;
  int bn = (int)(idx / D), d = (int)(idx % D);
  const float* e = enc + (size_t)(bn / NV) * S * D + d;
  float s = 0.f;
  for (int j = 0; j < S; ++j) s += e[(size_t)j * D];
  c[idx] = temb[idx] + s / (float)S;
}
__global__ void k_mv_fwd(const float* __restrict__ in, float* __restrict__ out,
                         int Bv, int NV, int L, int D) {
  long idx = (long)blockIdx.x * 256 + threadIdx.x;
  if (idx >= (long)Bv * NV * L * D) return;
  int d = (int)(idx % D);
  long r = idx / D;
  int n = (int)(r % NV); r /= NV;
  int l = (int)(r % L);
  int b = (int)(r / L);
  out[idx] = in[(((size_t)b * NV + n) * L + l) * D + d];
}
__global__ void k_mv_bwd(const float* __restrict__ in, float* __restrict__ out,
                         int Bv, int NV, int L, int D) {
  long idx = (long)blockIdx.x * 256 + threadIdx.x;
  if (idx >= (long)Bv * NV * L * D) return;
  int d = (int)(idx % D);
  long r = idx / D;
  int l = (int)(r % L); r /= L;
  int n = (int)(r % NV);
  int b = (int)(r / NV);
  out[idx] = in[(((size_t)b * L + l) * NV + n) * D + d];
}
__global__ void k_repeat_ref(const float* __restrict__ in, float* __restrict__ out,
                             int Bv, int NV, int L, int D) {
  long idx = (long)blockIdx.x * 256 + threadIdx.x;
  if (idx >= (long)Bv * NV * L * D) return;
  int d = (int)(idx % D);
  long r = idx / D;
  int l = (int)(r % L);
  int bn = (int)(r / L);
  out[idx] = in[((size_t)(bn / NV) * L + l) * D + d];
}
__global__ void k_patchify(const float* __restrict__ in, float* __restrict__ out,
                           int B, int C, int H, int Wd, int p) {
  int hp = H / p, wp = Wd / p, Cpp = C * p * p;
  long total = (long)B * hp * wp * Cpp;
  long idx = (long)blockIdx.x * 256 + threadIdx.x;
  if (idx >= total) return;
  int e = (int)(idx % Cpp);
  long r = idx / Cpp;
  int pw = (int)(r % wp); r /= wp;
  int ph = (int)(r % hp);
  int b = (int)(r / hp);
  int ix = e % p, iy = (e / p) % p, c = e / (p * p);
  out[idx] = in[(((size_t)b * C + c) * H + (size_t)ph * p + iy) * Wd + (size_t)pw * p + ix];
}
__global__ void k_tochw(const float* __restrict__ in, float* __restrict__ out,
                        int B, int h, int D) {
  long idx = (long)blockIdx.x * 256 + threadIdx.x;
  if (idx >= (long)B * D * h * h) return;
  int x = (int)(idx % h);
  long r = idx / h;
  int y = (int)(r % h); r /= h;
  int d = (int)(r % D);
  int b = (int)(r / D);
  out[idx] = in[((size_t)b * h * h + (size_t)y * h + x) * D + d];
}
__global__ void k_addpos(float* __restrict__ x, const float* __restrict__ pos, long n, long per) {
  long i = (long)blockIdx.x * 256 + threadIdx.x;
  if (i < n) x[i] += pos[i % per];
}
__global__ void k_unpatch(const float* __restrict__ y, float* __restrict__ out, int n) {
  long idx = (long)blockIdx.x * 256 + threadIdx.x;
  if (idx >= (long)n * 8 * 32 * 32) return;
  int X = (int)(idx % 32);
  long r = idx / 32;
  int Y = (int)(r % 32); r /= 32;
  int ch = (int)(r % 8);
  int b = (int)(r / 8);
  int hh = Y >> 1, pp = Y & 1, ww = X >> 1, qq = X & 1;
  out[idx] = y[((size_t)b * 256 + (size_t)hh * 16 + ww) * 32 + pp * 16 + qq * 8 + ch];
}

struct QS { int8_t* qa; float* adq; int8_t* qw; float* wdq; };

static void bitlinear_h(hipStream_t st, const float* X, int M, int K, const float* g,
                        const float* w, int N, const float* bias, float* Y, const QS& q) {
  k_rmsnorm_quant<<<dim3(M), dim3(256), 0, st>>>(X, g, q.qa, q.adq, M, K);
  long nw = (long)N * K;
  k_absmean<<<dim3(1), dim3(1024), 0, st>>>(w, nw, q.wdq);
  k_ternarize<<<dim3((unsigned)CDIV(nw, 256)), dim3(256), 0, st>>>(w, q.wdq, q.qw, nw);
  if ((M % 64 == 0) && (N % 128 == 0) && (K % 64 == 0)) {
    k_gemm_iu8_fast<<<dim3(N / 128, M / 64), dim3(256), 0, st>>>(q.qa, q.adq, q.qw, q.wdq,
                                                                 bias, Y, M, N, K);
  } else {
    k_gemm_iu8<<<dim3(CDIV(N, 64), CDIV(M, 64)), dim3(256), 0, st>>>(q.qa, q.adq, q.qw, q.wdq,
                                                                     bias, Y, M, N, K);
  }
}

static void hgrn_h(hipStream_t st, const float* X, int Bn, int Lh, const float* const* P,
                   float* out, float* t1, float* t2, float* t3, const QS& q) {
  const int D = 768;
  int M = Bn * Lh;
  long n = (long)M * D;
  bitlinear_h(st, X, M, D, P[1], P[0], D, nullptr, t1, q);
  bitlinear_h(st, X, M, D, P[3], P[2], D, nullptr, t2, q);
  k_gate<<<dim3((unsigned)CDIV(n, 256)), dim3(256), 0, st>>>(t1, t2, n);
  k_scan<<<dim3(CDIV(Bn * D, 256)), dim3(256), 0, st>>>(t2, t1, t3, Bn, Lh, D);
  bitlinear_h(st, X, M, D, P[5], P[4], D, nullptr, t1, q);
  k_headgate<<<dim3(CDIV(M * 12, 256)), dim3(256), 0, st>>>(t3, t1, P[8], t2, M, 12, 64);
  bitlinear_h(st, t2, M, D, P[7], P[6], D, nullptr, out, q);
}

extern "C" void kernel_launch(void* const* d_in, const int* in_sizes, int n_in,
                              void* d_out, int out_size, void* d_ws, size_t ws_size,
                              hipStream_t stream) {
  (void)in_sizes; (void)n_in; (void)out_size; (void)ws_size;
  hipStream_t st = stream;
  auto F = [&](int i) { return (const float*)d_in[i]; };

  constexpr int Bv = 2, NV = 4, BN = 8, L = 256, D = 768, INTER = 3072;
  const float* xin  = F(0);
  const float* tin  = F(1);
  const float* enc  = F(2);
  const float* cond = F(3);
  const float* ref  = F(4);

  char* wsb = (char*)d_ws;
  size_t off = 0;
  auto alloc = [&](size_t bytes) -> void* {
    void* p = wsb + off;
    off = (off + bytes + 255) & ~(size_t)255;
    return p;
  };
  const size_t MD = (size_t)BN * L * D;
  float* x_emb = (float*)alloc(MD * 4);
  float* xm    = (float*)alloc(MD * 4);
  float* a0    = (float*)alloc(MD * 4);
  float* a1    = (float*)alloc(MD * 4);
  float* a2    = (float*)alloc(MD * 4);
  float* a3    = (float*)alloc(MD * 4);
  float* t1b   = (float*)alloc(MD * 4);
  float* t2b   = (float*)alloc(MD * 4);
  float* t3b   = (float*)alloc(MD * 4);
  float* refbuf= (float*)alloc(MD * 4);
  float* big   = (float*)alloc((size_t)2048 * 2 * INTER * 4);
  float* hbuf  = (float*)alloc((size_t)2048 * INTER * 4);
  float* spA   = (float*)alloc((size_t)8192 * D * 4);
  float* spB   = (float*)alloc((size_t)8192 * D * 4);
  float* xpat  = (float*)alloc((size_t)2048 * 16 * 4);
  float* tf    = (float*)alloc((size_t)BN * 256 * 4);
  float* te1   = (float*)alloc((size_t)BN * D * 4);
  float* te2   = (float*)alloc((size_t)BN * D * 4);
  float* cbuf  = (float*)alloc((size_t)BN * D * 4);
  float* csilu = (float*)alloc((size_t)BN * D * 4);
  float* adaln = (float*)alloc((size_t)BN * 6 * D * 4);
  float* fadaln= (float*)alloc((size_t)BN * 2 * D * 4);
  float* ybuf  = (float*)alloc((size_t)2048 * 32 * 4);
  QS q;
  q.qa  = (int8_t*)alloc((size_t)2048 * INTER);
  q.adq = (float*)alloc((size_t)2048 * 4);
  q.qw  = (int8_t*)alloc((size_t)(2 * INTER) * D);
  q.wdq = (float*)alloc(256);

  k_patchify<<<dim3(CDIV(2048 * 16, 256)), dim3(256), 0, st>>>(xin, xpat, BN, 4, 32, 32, 2);
  k_gemm_f32<<<dim3((unsigned)CDIV((long)2048 * D, 256)), dim3(256), 0, st>>>(
      xpat, F(6), F(7), x_emb, 2048, D, 16);
  k_addpos<<<dim3((unsigned)CDIV((long)MD, 256)), dim3(256), 0, st>>>(x_emb, F(8), (long)MD,
                                                                      (long)L * D);

  k_tfreq<<<dim3(CDIV(BN * 128, 256)), dim3(256), 0, st>>>(tin, tf, BN);
  bitlinear_h(st, tf, BN, 256, F(10), F(9), D, F(11), te1, q);
  k_silu<<<dim3(CDIV(BN * D, 256)), dim3(256), 0, st>>>(te1, te1, (long)BN * D);
  bitlinear_h(st, te1, BN, D, F(13), F(12), D, F(14), te2, q);
  k_cond<<<dim3(CDIV(BN * D, 256)), dim3(256), 0, st>>>(enc, te2, cbuf, BN, NV, 77, D);
  k_silu<<<dim3(CDIV(BN * D, 256)), dim3(256), 0, st>>>(cbuf, csilu, (long)BN * D);

  {
    const int sb = 17;
    k_patchify<<<dim3(CDIV(8192 * 3, 256)), dim3(256), 0, st>>>(cond, hbuf, BN, 3, 32, 32, 1);
    k_gemm_f32<<<dim3((unsigned)CDIV((long)8192 * D, 256)), dim3(256), 0, st>>>(
        hbuf, F(15), F(16), spA, 8192, D, 3);
    k_lnmod<<<dim3(8192), dim3(256), 0, st>>>(spA, spB, 8192, D, 1, nullptr, 0, 0, 0,
                                              F(sb + 0), F(sb + 1), 1e-5f);
    for (int ch = 0; ch < 4; ++ch) {
      const float* xc = spB + (size_t)ch * 2048 * D;
      bitlinear_h(st, xc, 2048, D, F(sb + 3), F(sb + 2), 2 * INTER, nullptr, big, q);
      k_swiglu<<<dim3((unsigned)CDIV((long)2048 * INTER, 256)), dim3(256), 0, st>>>(big, hbuf,
                                                                                   2048, INTER);
      bitlinear_h(st, hbuf, 2048, INTER, F(sb + 5), F(sb + 4), D, nullptr,
                  spA + (size_t)ch * 2048 * D, q);
    }
    k_tochw<<<dim3((unsigned)CDIV((long)8192 * D, 256)), dim3(256), 0, st>>>(spA, spB, BN, 32, D);
    k_patchify<<<dim3((unsigned)CDIV((long)2048 * 3072, 256)), dim3(256), 0, st>>>(
        spB, hbuf, BN, D, 32, 32, 2);
    k_gemm_f32<<<dim3((unsigned)CDIV((long)2048 * D, 256)), dim3(256), 0, st>>>(
        hbuf, F(sb + 6), F(sb + 7), a0, 2048, D, 3072);
    k_addinp<<<dim3((unsigned)CDIV((long)MD, 256)), dim3(256), 0, st>>>(x_emb, a0, (long)MD);
  }

  k_repeat_ref<<<dim3((unsigned)CDIV((long)MD, 256)), dim3(256), 0, st>>>(ref, refbuf, Bv, NV, L, D);

  const unsigned g1 = (unsigned)CDIV((long)MD, 256);
  for (int blk = 0; blk < 4; ++blk) {
    const int b0 = 41 + blk * 34;
    const float* selfP[9]; const float* mvP[9]; const float* refP[9];
    for (int j = 0; j < 9; ++j) {
      selfP[j] = F(b0 + 3 + j);
      mvP[j]   = F(b0 + 12 + j);
      refP[j]  = F(b0 + 21 + j);
    }
    bitlinear_h(st, csilu, BN, D, F(b0 + 1), F(b0 + 0), 6 * D, F(b0 + 2), adaln, q);
    k_lnmod<<<dim3(2048), dim3(256), 0, st>>>(x_emb, xm, 2048, D, L, adaln, 6 * D, 0, D,
                                              nullptr, nullptr, 1e-6f);
    hgrn_h(st, xm, BN, L, selfP, a0, t1b, t2b, t3b, q);
    k_mv_fwd<<<dim3(g1), dim3(256), 0, st>>>(xm, a1, Bv, NV, L, D);
    hgrn_h(st, a1, Bv * L, NV, mvP, a2, t1b, t2b, t3b, q);
    k_mv_bwd<<<dim3(g1), dim3(256), 0, st>>>(a2, a3, Bv, NV, L, D);
    k_addinp<<<dim3(g1), dim3(256), 0, st>>>(a0, a3, (long)MD);
    hgrn_h(st, refbuf, BN, L, refP, a3, t1b, t2b, t3b, q);
    k_addinp<<<dim3(g1), dim3(256), 0, st>>>(a0, a3, (long)MD);
    k_resmod<<<dim3(g1), dim3(256), 0, st>>>(x_emb, a0, adaln, 6 * D, 2 * D, 2048, D, L);
    k_lnmod<<<dim3(2048), dim3(256), 0, st>>>(x_emb, xm, 2048, D, L, adaln, 6 * D, 3 * D,
                                              4 * D, nullptr, nullptr, 1e-6f);
    bitlinear_h(st, xm, 2048, D, F(b0 + 31), F(b0 + 30), 2 * INTER, nullptr, big, q);
    k_swiglu<<<dim3((unsigned)CDIV((long)2048 * INTER, 256)), dim3(256), 0, st>>>(big, hbuf,
                                                                                 2048, INTER);
    bitlinear_h(st, hbuf, 2048, INTER, F(b0 + 33), F(b0 + 32), D, nullptr, a0, q);
    k_resmod<<<dim3(g1), dim3(256), 0, st>>>(x_emb, a0, adaln, 6 * D, 5 * D, 2048, D, L);
  }

  bitlinear_h(st, csilu, BN, D, F(178), F(177), 2 * D, F(179), fadaln, q);
  k_lnmod<<<dim3(2048), dim3(256), 0, st>>>(x_emb, xm, 2048, D, L, fadaln, 2 * D, 0, D,
                                            nullptr, nullptr, 1e-6f);
  bitlinear_h(st, xm, 2048, D, F(181), F(180), 32, F(182), ybuf, q);
  k_unpatch<<<dim3((unsigned)CDIV((long)BN * 8 * 32 * 32, 256)), dim3(256), 0, st>>>(
      ybuf, (float*)d_out, BN);
}
